// GCNEncoder_28965259444843
// MI455X (gfx1250) — matrix operand order repaired and checked by reading, not compiled, not run on hardware
//
#include <hip/hip_runtime.h>

// ---------- types ----------
typedef __attribute__((ext_vector_type(16))) __bf16 v16bf;
typedef __attribute__((ext_vector_type(8)))  float  v8f;
typedef __attribute__((ext_vector_type(4)))  unsigned int u32x4;
typedef __attribute__((ext_vector_type(4)))  float  f32x4;
typedef __attribute__((ext_vector_type(4)))  unsigned short u16x4;
typedef unsigned short u16;

union FragBF {            // 16 bf16 elements = one WMMA A/B operand (8 VGPRs)
  v16bf v;
  u32x4 q[2];
};

__device__ __forceinline__ u16 f32_to_bf16(float f) {
  unsigned u = __float_as_uint(f);
  u += 0x7FFFu + ((u >> 16) & 1u);       // round-to-nearest-even
  return (u16)(u >> 16);
}

__device__ __forceinline__ void atom_add_f32(float* p, float v) {
  unsafeAtomicAdd(p, v);                 // global_atomic_add_f32, no CAS loop
}

// ---------- conversion / prep kernels ----------
// 4 elements per thread: b128 load -> b64 store. n must be a multiple of 4 (Nn*128 is).
__global__ __launch_bounds__(256) void cvt_f32_to_bf16_x4(const float* __restrict__ src,
                                                          u16* __restrict__ dst, int n4) {
  int i = blockIdx.x * 256 + threadIdx.x;
  if (i >= n4) return;
  f32x4 v = *(const f32x4*)(src + (size_t)i * 4);
  u16x4 o;
  o.x = f32_to_bf16(v.x); o.y = f32_to_bf16(v.y);
  o.z = f32_to_bf16(v.z); o.w = f32_to_bf16(v.w);
  *(u16x4*)(dst + (size_t)i * 4) = o;
}

// Build transposed bf16 weights: W1t[n*128+k] = W1[k*128+n];
// Wct packs [W_mu | W_ls] -> 128 output cols, transposed to [n][k].
__global__ __launch_bounds__(256) void prep_weights(const float* __restrict__ W1,
                                                    const float* __restrict__ Wmu,
                                                    const float* __restrict__ Wls,
                                                    u16* __restrict__ W1t,
                                                    u16* __restrict__ Wct) {
  int tid = blockIdx.x * 256 + threadIdx.x;     // 0 .. 16383
  if (tid >= 128 * 128) return;
  int k = tid & 127;
  int n = tid >> 7;
  W1t[(size_t)n * 128 + k] = f32_to_bf16(W1[(size_t)k * 128 + n]);
  float w = (n < 64) ? Wmu[(size_t)k * 64 + n] : Wls[(size_t)k * 64 + (n - 64)];
  Wct[(size_t)n * 128 + k] = f32_to_bf16(w);
}

// ---------- degree / normalization ----------
__global__ __launch_bounds__(256) void deg_init(float* __restrict__ deg, int n) {
  int i = blockIdx.x * 256 + threadIdx.x;
  if (i < n) deg[i] = 1.0f;                      // self-loop
}

__global__ __launch_bounds__(256) void deg_count(const int* __restrict__ cols,
                                                 float* __restrict__ deg, int E) {
  int e = blockIdx.x * 256 + threadIdx.x;
  if (e < E) atom_add_f32(&deg[cols[e]], 1.0f);
}

__global__ __launch_bounds__(256) void deg_rsqrt(float* __restrict__ deg, int n) {
  int i = blockIdx.x * 256 + threadIdx.x;
  if (i < n) deg[i] = rsqrtf(deg[i]);            // deg >= 1 always
}

// ---------- WMMA GEMM: C[M,128] = A[M,128](bf16) * Bt[128,128](bf16, pre-transposed) ----------
// One wave32 per 16x16 output tile; K=128 in 4 steps of v_wmma_f32_16x16x32_bf16.
// REQUIRES M % 16 == 0 (true here: M = 100000 = 6250*16) -> no tail path at all.
__global__ __launch_bounds__(256) void gemm_bf16_128(const u16* __restrict__ Ab,
                                                     const u16* __restrict__ Bt,
                                                     float* __restrict__ C,
                                                     int M) {
  const int K = 128;
  int wave = (blockIdx.x * 256 + (int)threadIdx.x) >> 5;
  int lane = threadIdx.x & 31;
  int mtiles = M >> 4;
  if (wave >= mtiles * 8) return;
  int mt = wave >> 3;
  int nt = wave & 7;
  int m0 = mt << 4, n0 = nt << 4;

  int half = lane >> 4;          // 0: lanes 0-15, 1: lanes 16-31
  int l15  = lane & 15;

  const u16* __restrict__ Arow = Ab + ((size_t)(m0 + l15) << 7);
  const u16* __restrict__ Bcol = Bt + ((size_t)(n0 + l15) << 7);

  v8f acc = {};
  #pragma unroll
  for (int k0 = 0; k0 < K; k0 += 32) {
    FragBF a, b;
    // A (16x32, MxK): lane half selects K sub-block {0..7,16..23} vs {8..15,24..31}
    int kA = k0 + half * 8;
    a.q[0] = *(const u32x4*)(Arow + kA);
    a.q[1] = *(const u32x4*)(Arow + kA + 16);
    // B (32x16, KxN): lane half selects K sub-block {0..7,16..23} vs {8..15,24..31}
    int kB = k0 + half * 8;
    b.q[0] = *(const u32x4*)(Bcol + kB);
    b.q[1] = *(const u32x4*)(Bcol + kB + 16);
    acc = __builtin_amdgcn_wmma_f32_16x16x32_bf16(
        /*neg_a=*/false, a.v, /*neg_b=*/false, b.v,
        /*c_mod=*/(short)0, acc, /*reuse_a=*/false, /*reuse_b=*/false);
  }

  // C/D layout: VGPR r -> row m0 + half*8 + r, col n0 + (lane&15)
  float* __restrict__ Cout = C + ((size_t)(m0 + half * 8) << 7) + n0 + l15;
  #pragma unroll
  for (int r = 0; r < 8; ++r) Cout[(size_t)r << 7] = acc[r];
}

// ---------- edge aggregation: one wave per edge, 128 channels ----------
__global__ __launch_bounds__(256) void agg_edges128(const int* __restrict__ rows,
                                                    const int* __restrict__ cols,
                                                    const float* __restrict__ dinv,
                                                    const float* __restrict__ pre,
                                                    float* __restrict__ acc, int E) {
  int gid = blockIdx.x * 256 + (int)threadIdx.x;
  int e = gid >> 5;
  if (e >= E) return;
  int lane = threadIdx.x & 31;
  int r = rows[e], c = cols[e];
  float nrm = dinv[r] * dinv[c];
  const f32x4 v = *(const f32x4*)(pre + ((size_t)r << 7) + lane * 4);
  float* dst = acc + ((size_t)c << 7) + lane * 4;
  atom_add_f32(dst + 0, v.x * nrm);
  atom_add_f32(dst + 1, v.y * nrm);
  atom_add_f32(dst + 2, v.z * nrm);
  atom_add_f32(dst + 3, v.w * nrm);
}

// ---------- epilogue 1: h = relu(acc + self_loop + b1) -> bf16 ----------
__global__ __launch_bounds__(256) void epilogue1(const float* __restrict__ acc,
                                                 const float* __restrict__ pre,
                                                 const float* __restrict__ dinv,
                                                 const float* __restrict__ b1,
                                                 u16* __restrict__ hb, int total) {
  int i = blockIdx.x * 256 + threadIdx.x;
  if (i >= total) return;
  int node = i >> 7;
  int ch   = i & 127;
  float di = dinv[node];
  float v = acc[i] + pre[i] * (di * di) + b1[ch];
  v = fmaxf(v, 0.0f);
  hb[i] = f32_to_bf16(v);
}

// ---------- epilogue 2: split packed [mu|logstd] into d_out ----------
__global__ __launch_bounds__(256) void epilogue2(const float* __restrict__ acc,
                                                 const float* __restrict__ pre,
                                                 const float* __restrict__ dinv,
                                                 const float* __restrict__ bmu,
                                                 const float* __restrict__ bls,
                                                 float* __restrict__ out,
                                                 int Nn, int total) {
  int i = blockIdx.x * 256 + threadIdx.x;
  if (i >= total) return;
  int node = i >> 7;
  int ch   = i & 127;
  float di = dinv[node];
  float v = acc[i] + pre[i] * (di * di);
  if (ch < 64) {
    out[(size_t)node * 64 + ch] = v + bmu[ch];                                // mu
  } else {
    out[(size_t)Nn * 64 + (size_t)node * 64 + (ch - 64)] = v + bls[ch - 64];  // logstd
  }
}

// ---------- host ----------
extern "C" void kernel_launch(void* const* d_in, const int* in_sizes, int n_in,
                              void* d_out, int out_size, void* d_ws, size_t ws_size,
                              hipStream_t stream) {
  const float* x    = (const float*)d_in[0];
  const int*   eidx = (const int*)d_in[1];
  const float* W1   = (const float*)d_in[2];
  const float* b1   = (const float*)d_in[3];
  const float* Wmu  = (const float*)d_in[4];
  const float* bmu  = (const float*)d_in[5];
  const float* Wls  = (const float*)d_in[6];
  const float* bls  = (const float*)d_in[7];
  float* out = (float*)d_out;

  const int Nn = in_sizes[0] / 128;   // 100000 (multiple of 16)
  const int E  = in_sizes[1] / 2;     // 1600000
  const int*   rows = eidx;           // sources
  const int*   cols = eidx + E;       // targets

  // workspace layout
  char* ws = (char*)d_ws;
  size_t off = 0;
  auto take = [&](size_t bytes) { size_t o = off; off += (bytes + 255) & ~(size_t)255; return o; };
  u16*   xb   = (u16*)  (ws + take((size_t)Nn * 128 * sizeof(u16)));   // xb, later hb
  float* pre  = (float*)(ws + take((size_t)Nn * 128 * sizeof(float))); // pre1, later pre2
  float* acc  = (float*)(ws + take((size_t)Nn * 128 * sizeof(float)));
  float* dinv = (float*)(ws + take((size_t)Nn * sizeof(float)));
  u16*   W1t  = (u16*)  (ws + take((size_t)128 * 128 * sizeof(u16)));
  u16*   Wct  = (u16*)  (ws + take((size_t)128 * 128 * sizeof(u16)));
  (void)ws_size;

  const int total = Nn * 128;
  const int B = 256;
  int blkTotal = (total + B - 1) / B;
  int blkCvt   = (total / 4 + B - 1) / B;
  int blkN     = (Nn + B - 1) / B;
  int blkE     = (E + B - 1) / B;
  int gemmWaves = (Nn >> 4) * 8;
  int blkGemm  = (gemmWaves * 32 + B - 1) / B;
  long long aggThreads = (long long)E * 32;
  int blkAgg   = (int)((aggThreads + B - 1) / B);

  // prep: bf16 inputs, transposed/packed weights, degrees
  cvt_f32_to_bf16_x4<<<blkCvt, B, 0, stream>>>(x, xb, total / 4);
  prep_weights<<<(128 * 128 + B - 1) / B, B, 0, stream>>>(W1, Wmu, Wls, W1t, Wct);
  deg_init<<<blkN, B, 0, stream>>>(dinv, Nn);
  deg_count<<<blkE, B, 0, stream>>>(cols, dinv, E);
  deg_rsqrt<<<blkN, B, 0, stream>>>(dinv, Nn);

  // layer 1: pre1 = x @ W1 (WMMA), aggregate, relu -> hb (bf16)
  gemm_bf16_128<<<blkGemm, B, 0, stream>>>(xb, W1t, pre, Nn);
  hipMemsetAsync(acc, 0, (size_t)total * sizeof(float), stream);
  agg_edges128<<<blkAgg, B, 0, stream>>>(rows, cols, dinv, pre, acc, E);
  epilogue1<<<blkTotal, B, 0, stream>>>(acc, pre, dinv, b1, xb /*reuse as hb*/, total);

  // layer 2 (mu & logstd fused): pre2 = h @ [Wmu|Wls], aggregate, split to d_out
  gemm_bf16_128<<<blkGemm, B, 0, stream>>>(xb, Wct, pre, Nn);
  hipMemsetAsync(acc, 0, (size_t)total * sizeof(float), stream);
  agg_edges128<<<blkAgg, B, 0, stream>>>(rows, cols, dinv, pre, acc, E);
  epilogue2<<<blkTotal, B, 0, stream>>>(acc, pre, dinv, bmu, bls, out, Nn, total);
}
